// SE3TransformerBaseline_56873956934481
// MI455X (gfx1250) — matrix operand order repaired and checked by reading, not compiled, not run on hardware
//
#include <hip/hip_runtime.h>

// ---------------------------------------------------------------------------
// Types / helpers
// ---------------------------------------------------------------------------
typedef __attribute__((ext_vector_type(16))) __bf16 v16bf;
typedef __attribute__((ext_vector_type(8)))  __bf16 v8bf;
typedef __attribute__((ext_vector_type(8)))  float  v8f;
typedef unsigned short u16;
typedef __attribute__((ext_vector_type(8)))  unsigned short v8u16;

#define MODE_BF16    0
#define MODE_BF16T   1
#define MODE_ACC     2
#define MODE_GELU    3
#define MODE_F32BIAS 4

#define TOK   2048      // B*N
#define NTOK  1024      // N
#define DMODEL 512
#define NHEAD 8
#define DHEAD 64

__device__ __forceinline__ u16 f2bf(float f) {
  unsigned u = __float_as_uint(f);
  u += 0x7FFFu + ((u >> 16) & 1u);          // round-to-nearest-even
  return (u16)(u >> 16);
}
__device__ __forceinline__ v8bf ld8(const u16* p) {
  return *reinterpret_cast<const v8bf*>(p);
}
__device__ __forceinline__ v16bf mk16(v8bf lo, v8bf hi) {
  v16bf r;
#pragma unroll
  for (int i = 0; i < 8; ++i) { r[i] = lo[i]; r[i + 8] = hi[i]; }
  return r;
}
__device__ __forceinline__ v8f wmma_bf16(v16bf a, v16bf b, v8f c) {
  // D = A(16x32 bf16) * B(32x16 bf16) + C(16x16 f32)
  return __builtin_amdgcn_wmma_f32_16x16x32_bf16(false, a, false, b, (short)0, c,
                                                 false, false);
}
__device__ __forceinline__ float gelu_f(float x) {
  float t = tanhf(0.7978845608028654f * (x + 0.044715f * x * x * x));
  return 0.5f * x * (1.0f + t);
}
__device__ __forceinline__ float redmax16(float v) {
#pragma unroll
  for (int m = 1; m < 16; m <<= 1) v = fmaxf(v, __shfl_xor(v, m, 32));
  return v;
}
__device__ __forceinline__ float redsum16(float v) {
#pragma unroll
  for (int m = 1; m < 16; m <<= 1) v += __shfl_xor(v, m, 32);
  return v;
}

// ---------------------------------------------------------------------------
// Weight transpose + f32->bf16 convert:  src [K][Nc] f32  ->  dst [Nc][K] bf16
// ---------------------------------------------------------------------------
__global__ void wt_kernel(const float* __restrict__ src, u16* __restrict__ dst,
                          int K, int Nc) {
  int idx = blockIdx.x * 256 + threadIdx.x;
  if (idx >= K * Nc) return;
  int k = idx / Nc, n = idx % Nc;
  dst[(size_t)n * K + k] = f2bf(src[idx]);
}

// ---------------------------------------------------------------------------
// Pairwise distances: dist[b][i][j]
// ---------------------------------------------------------------------------
__global__ void dist_kernel(const float* __restrict__ x, float* __restrict__ dist) {
  size_t idx = (size_t)blockIdx.x * 256 + threadIdx.x;  // over 2*1024*1024
  int b = (int)(idx >> 20);
  int i = (int)((idx >> 10) & 1023);
  int j = (int)(idx & 1023);
  const float* xi = x + ((size_t)b * NTOK + i) * 3;
  const float* xj = x + ((size_t)b * NTOK + j) * 3;
  float dx = xi[0] - xj[0], dy = xi[1] - xj[1], dz = xi[2] - xj[2];
  dist[idx] = sqrtf(dx * dx + dy * dy + dz * dz + 1e-8f);
}

// ---------------------------------------------------------------------------
// Radial bias (per layer): rbias[b][h][i][j] = sum_m tanh(d*w1[m])*w2[m][h]
// masked entries set to -1e9 (radius cutoff).
// ---------------------------------------------------------------------------
__global__ void rbias_kernel(const float* __restrict__ dist,
                             const float* __restrict__ w1,   // [32]
                             const float* __restrict__ w2,   // [32][8]
                             float* __restrict__ rb) {
  size_t idx = (size_t)blockIdx.x * 256 + threadIdx.x;  // over 2*1024*1024
  float d = dist[idx];
  size_t b  = idx >> 20;
  size_t ij = idx & ((1u << 20) - 1);
  float acc[NHEAD];
#pragma unroll
  for (int h = 0; h < NHEAD; ++h) acc[h] = 0.0f;
  if (d <= 10.0f) {
#pragma unroll 4
    for (int m = 0; m < 32; ++m) {
      float t = tanhf(d * w1[m]);
#pragma unroll
      for (int h = 0; h < NHEAD; ++h) acc[h] += t * w2[m * NHEAD + h];
    }
  } else {
#pragma unroll
    for (int h = 0; h < NHEAD; ++h) acc[h] = -1e9f;
  }
#pragma unroll
  for (int h = 0; h < NHEAD; ++h)
    rb[((b * NHEAD + h) << 20) + ij] = acc[h];
}

// ---------------------------------------------------------------------------
// LayerNorm (parameter-free) f32 -> bf16, or plain convert when normalize==0
// one block (256 thr) per row of 512
// ---------------------------------------------------------------------------
__global__ void ln_kernel(const float* __restrict__ src, u16* __restrict__ dst,
                          int normalize) {
  int row = blockIdx.x, t = threadIdx.x;
  const float* r = src + (size_t)row * DMODEL;
  float a = r[t], b = r[t + 256];
  if (!normalize) {
    dst[(size_t)row * DMODEL + t] = f2bf(a);
    dst[(size_t)row * DMODEL + t + 256] = f2bf(b);
    return;
  }
  __shared__ float red[256];
  red[t] = a + b;
  __syncthreads();
  for (int o = 128; o > 0; o >>= 1) { if (t < o) red[t] += red[t + o]; __syncthreads(); }
  float mean = red[0] * (1.0f / DMODEL);
  __syncthreads();
  float da = a - mean, db = b - mean;
  red[t] = da * da + db * db;
  __syncthreads();
  for (int o = 128; o > 0; o >>= 1) { if (t < o) red[t] += red[t + o]; __syncthreads(); }
  float rstd = rsqrtf(red[0] * (1.0f / DMODEL) + 1e-5f);
  dst[(size_t)row * DMODEL + t] = f2bf(da * rstd);
  dst[(size_t)row * DMODEL + t + 256] = f2bf(db * rstd);
}

// ---------------------------------------------------------------------------
// Generic bf16 WMMA GEMM: one wave computes a 16x64 tile of C = A[M,K] * Wt^T
//   A: bf16 row-major [M][K];  Wt: bf16 [Nc][K] (pre-transposed weights)
// grid = (M/16, Nc/64), block = 32
// Epilogue selected by mode.
// ---------------------------------------------------------------------------
__global__ void gemm_kernel(const u16* __restrict__ A, const u16* __restrict__ Wt,
                            int K, int Nc,
                            u16* __restrict__ outb, float* __restrict__ accum,
                            const float* __restrict__ bias, int mode) {
  int mb = blockIdx.x * 16;
  int nb = blockIdx.y * 64;
  int lane = threadIdx.x;
  int n  = lane & 15;
  int hi = lane >> 4;

  v8f C0 = {}, C1 = {}, C2 = {}, C3 = {};
  const u16* Arow = A + (size_t)(mb + n) * K;
  const u16* W0 = Wt + (size_t)(nb + 0 * 16 + n) * K;
  const u16* W1 = Wt + (size_t)(nb + 1 * 16 + n) * K;
  const u16* W2 = Wt + (size_t)(nb + 2 * 16 + n) * K;
  const u16* W3 = Wt + (size_t)(nb + 3 * 16 + n) * K;

  for (int kc = 0; kc < K; kc += 32) {
    // A fragment: 16 contiguous K values per lane, K = kc+16*hi+..
    v16bf a = mk16(ld8(Arow + kc + 16 * hi), ld8(Arow + kc + 16 * hi + 8));
    int ko = kc + 16 * hi;  // B fragment: 16 contiguous K values per lane
    C0 = wmma_bf16(a, mk16(ld8(W0 + ko), ld8(W0 + ko + 8)), C0);
    C1 = wmma_bf16(a, mk16(ld8(W1 + ko), ld8(W1 + ko + 8)), C1);
    C2 = wmma_bf16(a, mk16(ld8(W2 + ko), ld8(W2 + ko + 8)), C2);
    C3 = wmma_bf16(a, mk16(ld8(W3 + ko), ld8(W3 + ko + 8)), C3);
  }

  v8f* Cs[4] = { &C0, &C1, &C2, &C3 };
  if (mode == MODE_BF16 || mode == MODE_GELU) {
#pragma unroll
    for (int t = 0; t < 4; ++t)
#pragma unroll
      for (int r = 0; r < 8; ++r) {
        float v = (*Cs[t])[r];
        if (mode == MODE_GELU) v = gelu_f(v);
        outb[(size_t)(mb + r + 8 * hi) * Nc + nb + t * 16 + n] = f2bf(v);
      }
  } else if (mode == MODE_BF16T) {
    // store V transposed per batch: out[b][feature][token]
    int b = mb >> 10;
    int jbase = (mb & 1023) + 8 * hi;
#pragma unroll
    for (int t = 0; t < 4; ++t) {
      v8u16 pk;
#pragma unroll
      for (int r = 0; r < 8; ++r) pk[r] = f2bf((*Cs[t])[r]);
      *reinterpret_cast<v8u16*>(
          outb + ((size_t)(b * DMODEL + nb + t * 16 + n)) * NTOK + jbase) = pk;
    }
  } else if (mode == MODE_ACC) {
#pragma unroll
    for (int t = 0; t < 4; ++t)
#pragma unroll
      for (int r = 0; r < 8; ++r) {
        size_t idx = (size_t)(mb + r + 8 * hi) * Nc + nb + t * 16 + n;
        accum[idx] += (*Cs[t])[r];
      }
  } else {  // MODE_F32BIAS
#pragma unroll
    for (int t = 0; t < 4; ++t)
#pragma unroll
      for (int r = 0; r < 8; ++r) {
        size_t idx = (size_t)(mb + r + 8 * hi) * Nc + nb + t * 16 + n;
        accum[idx] = (*Cs[t])[r] + bias[nb + t * 16 + n];
      }
  }
}

// ---------------------------------------------------------------------------
// Flash attention: one wave per (b, h, 16-row i-tile).
//  q,k: bf16 [tok][512] (head h at cols h*64..); vT: bf16 [b][512][1024]
//  rbias: f32 [b][h][i][j] (mask folded in); ob: bf16 [tok][512]
// ---------------------------------------------------------------------------
__global__ void attn_kernel(const u16* __restrict__ q, const u16* __restrict__ k,
                            const u16* __restrict__ vT,
                            const float* __restrict__ rbias,
                            u16* __restrict__ ob) {
  int it = blockIdx.x & 63;
  int bh = blockIdx.x >> 6;
  int b = bh >> 3, h = bh & 7;
  int lane = threadIdx.x;
  int n  = lane & 15;
  int hi = lane >> 4;

  __shared__ __align__(16) u16 Plds[16 * 32];

  const u16* qrow = q + ((size_t)(b * NTOK + it * 16 + n)) * DMODEL + h * DHEAD;
  v16bf Aq0 = mk16(ld8(qrow + 16 * hi),      ld8(qrow + 16 * hi + 8));
  v16bf Aq1 = mk16(ld8(qrow + 32 + 16 * hi), ld8(qrow + 32 + 16 * hi + 8));

  v8f O0 = {}, O1 = {}, O2 = {}, O3 = {};
  float rowmax[8], rowsum[8];
#pragma unroll
  for (int r = 0; r < 8; ++r) { rowmax[r] = -3.0e38f; rowsum[r] = 0.0f; }

  const float* rbrow = rbias + (((size_t)bh * NTOK + it * 16)) * NTOK;

  for (int j0 = 0; j0 < NTOK; j0 += 32) {
    // ---- S = Q * K^T  (16x32 as two 16x16 tiles) ----
    const u16* k0 = k + ((size_t)(b * NTOK + j0 + n)) * DMODEL + h * DHEAD;
    const u16* k1 = k0 + (size_t)16 * DMODEL;
    v8f S0 = {}, S1 = {};
    S0 = wmma_bf16(Aq0, mk16(ld8(k0 + 16 * hi),      ld8(k0 + 16 * hi + 8)), S0);
    S0 = wmma_bf16(Aq1, mk16(ld8(k0 + 32 + 16 * hi), ld8(k0 + 32 + 16 * hi + 8)), S0);
    S1 = wmma_bf16(Aq0, mk16(ld8(k1 + 16 * hi),      ld8(k1 + 16 * hi + 8)), S1);
    S1 = wmma_bf16(Aq1, mk16(ld8(k1 + 32 + 16 * hi), ld8(k1 + 32 + 16 * hi + 8)), S1);

    // ---- bias + mask + online softmax ----
    float corr[8];
#pragma unroll
    for (int r = 0; r < 8; ++r) {
      size_t ro = (size_t)(r + 8 * hi) * NTOK + j0 + n;
      float v0 = S0[r] * 0.125f + rbrow[ro];
      float v1 = S1[r] * 0.125f + rbrow[ro + 16];
      float mx = redmax16(fmaxf(v0, v1));
      float mnew = fmaxf(rowmax[r], mx);
      float c  = __expf(rowmax[r] - mnew);
      float p0 = __expf(v0 - mnew);
      float p1 = __expf(v1 - mnew);
      float ps = redsum16(p0 + p1);
      rowsum[r] = rowsum[r] * c + ps;
      rowmax[r] = mnew;
      corr[r] = c;
      Plds[(r + 8 * hi) * 32 + n]      = f2bf(p0);
      Plds[(r + 8 * hi) * 32 + 16 + n] = f2bf(p1);
    }
#pragma unroll
    for (int r = 0; r < 8; ++r) {
      O0[r] *= corr[r]; O1[r] *= corr[r]; O2[r] *= corr[r]; O3[r] *= corr[r];
    }
    __syncthreads();

    // ---- O += P * V  (P 16x32 bf16 from LDS; V chunk from transposed buffer)
    v16bf Pf = mk16(ld8(Plds + n * 32 + 16 * hi), ld8(Plds + n * 32 + 16 * hi + 8));
    const u16* vbase =
        vT + ((size_t)(b * DMODEL + h * DHEAD + n)) * NTOK + j0 + 16 * hi;
    O0 = wmma_bf16(Pf, mk16(ld8(vbase),               ld8(vbase + 8)), O0);
    O1 = wmma_bf16(Pf, mk16(ld8(vbase + 16 * NTOK),   ld8(vbase + 16 * NTOK + 8)), O1);
    O2 = wmma_bf16(Pf, mk16(ld8(vbase + 32 * NTOK),   ld8(vbase + 32 * NTOK + 8)), O2);
    O3 = wmma_bf16(Pf, mk16(ld8(vbase + 48 * NTOK),   ld8(vbase + 48 * NTOK + 8)), O3);
    __syncthreads();
  }

  u16* orow = ob + ((size_t)(b * NTOK + it * 16)) * DMODEL + h * DHEAD;
#pragma unroll
  for (int r = 0; r < 8; ++r) {
    float inv = 1.0f / rowsum[r];  // diagonal always inside radius -> rowsum > 0
    size_t ro = (size_t)(r + 8 * hi) * DMODEL;
    orow[ro + 0 * 16 + n] = f2bf(O0[r] * inv);
    orow[ro + 1 * 16 + n] = f2bf(O1[r] * inv);
    orow[ro + 2 * 16 + n] = f2bf(O2[r] * inv);
    orow[ro + 3 * 16 + n] = f2bf(O3[r] * inv);
  }
}

// ---------------------------------------------------------------------------
// Host launcher
// ---------------------------------------------------------------------------
extern "C" void kernel_launch(void* const* d_in, const int* in_sizes, int n_in,
                              void* d_out, int out_size, void* d_ws, size_t ws_size,
                              hipStream_t stream) {
  const float* h   = (const float*)d_in[0];
  const float* x   = (const float*)d_in[1];
  const float* Wq  = (const float*)d_in[2];
  const float* Wk  = (const float*)d_in[3];
  const float* Wv  = (const float*)d_in[4];
  const float* Wo  = (const float*)d_in[5];
  const float* Wr1 = (const float*)d_in[6];
  const float* Wr2 = (const float*)d_in[7];
  const float* Wf1 = (const float*)d_in[8];
  const float* Wf2 = (const float*)d_in[9];
  const float* Wp  = (const float*)d_in[10];
  const float* bp  = (const float*)d_in[11];
  float* out = (float*)d_out;
  char* ws = (char*)d_ws;

  const size_t MB = 1024 * 1024;
  float* dist  = (float*)(ws + 0 * MB);      //  8 MB  [2][1024][1024]
  float* rbias = (float*)(ws + 8 * MB);      // 64 MB  [2][8][1024][1024]
  float* hcur  = (float*)(ws + 72 * MB);     //  4 MB  [2048][512] f32
  u16* hn   = (u16*)(ws + 76 * MB);          //  2 MB
  u16* qb   = (u16*)(ws + 78 * MB);          //  2 MB
  u16* kb   = (u16*)(ws + 80 * MB);          //  2 MB
  u16* vtb  = (u16*)(ws + 82 * MB);          //  2 MB  [2][512][1024]
  u16* obuf = (u16*)(ws + 84 * MB);          //  2 MB
  u16* ffb  = (u16*)(ws + 86 * MB);          //  8 MB  [2048][2048]
  u16* WqT  = (u16*)(ws + 94 * MB);          //  8 MB  (4 layers x [512][512])
  u16* WkT  = (u16*)(ws + 102 * MB);
  u16* WvT  = (u16*)(ws + 110 * MB);
  u16* WoT  = (u16*)(ws + 118 * MB);
  u16* Wf1T = (u16*)(ws + 126 * MB);         //  8 MB  (4 x [2048][512])
  u16* Wf2T = (u16*)(ws + 134 * MB);         //  8 MB  (4 x [512][2048])
  u16* WpT  = (u16*)(ws + 142 * MB);         //  128 KB [128][512]

  const int D = DMODEL, FF = 4 * DMODEL;

  // ---- weight prep (transpose + bf16 convert) ----
  for (int l = 0; l < 4; ++l) {
    size_t sq = (size_t)l * D * D;
    wt_kernel<<<(D * D + 255) / 256, 256, 0, stream>>>(Wq + sq, WqT + sq, D, D);
    wt_kernel<<<(D * D + 255) / 256, 256, 0, stream>>>(Wk + sq, WkT + sq, D, D);
    wt_kernel<<<(D * D + 255) / 256, 256, 0, stream>>>(Wv + sq, WvT + sq, D, D);
    wt_kernel<<<(D * D + 255) / 256, 256, 0, stream>>>(Wo + sq, WoT + sq, D, D);
    size_t sf = (size_t)l * D * FF;
    wt_kernel<<<(D * FF + 255) / 256, 256, 0, stream>>>(Wf1 + sf, Wf1T + sf, D, FF);
    wt_kernel<<<(D * FF + 255) / 256, 256, 0, stream>>>(Wf2 + sf, Wf2T + sf, FF, D);
  }
  wt_kernel<<<(D * 128 + 255) / 256, 256, 0, stream>>>(Wp, WpT, D, 128);

  // ---- geometry ----
  dist_kernel<<<(2 * 1024 * 1024) / 256, 256, 0, stream>>>(x, dist);

  // ---- h working copy ----
  hipMemcpyAsync(hcur, h, (size_t)TOK * D * sizeof(float),
                 hipMemcpyDeviceToDevice, stream);

  for (int l = 0; l < 4; ++l) {
    rbias_kernel<<<(2 * 1024 * 1024) / 256, 256, 0, stream>>>(
        dist, Wr1 + (size_t)l * 32, Wr2 + (size_t)l * 32 * NHEAD, rbias);

    ln_kernel<<<TOK, 256, 0, stream>>>(hcur, hn, 1);

    size_t sq = (size_t)l * D * D;
    gemm_kernel<<<dim3(TOK / 16, D / 64), 32, 0, stream>>>(
        hn, WqT + sq, D, D, qb, nullptr, nullptr, MODE_BF16);
    gemm_kernel<<<dim3(TOK / 16, D / 64), 32, 0, stream>>>(
        hn, WkT + sq, D, D, kb, nullptr, nullptr, MODE_BF16);
    gemm_kernel<<<dim3(TOK / 16, D / 64), 32, 0, stream>>>(
        hn, WvT + sq, D, D, vtb, nullptr, nullptr, MODE_BF16T);

    attn_kernel<<<2 * NHEAD * (NTOK / 16), 32, 0, stream>>>(qb, kb, vtb, rbias, obuf);

    gemm_kernel<<<dim3(TOK / 16, D / 64), 32, 0, stream>>>(
        obuf, WoT + sq, D, D, nullptr, hcur, nullptr, MODE_ACC);

    ln_kernel<<<TOK, 256, 0, stream>>>(hcur, hn, 1);

    size_t sf = (size_t)l * D * FF;
    gemm_kernel<<<dim3(TOK / 16, FF / 64), 32, 0, stream>>>(
        hn, Wf1T + sf, D, FF, ffb, nullptr, nullptr, MODE_GELU);
    gemm_kernel<<<dim3(TOK / 16, D / 64), 32, 0, stream>>>(
        ffb, Wf2T + sf, FF, D, nullptr, hcur, nullptr, MODE_ACC);
  }

  // ---- final projection (no LN): out = h @ Wp + bp ----
  ln_kernel<<<TOK, 256, 0, stream>>>(hcur, hn, 0);
  gemm_kernel<<<dim3(TOK / 16, 128 / 64), 32, 0, stream>>>(
      hn, WpT, D, 128, nullptr, out, bp, MODE_F32BIAS);
}
